// SelfAttention_73864847556854
// MI455X (gfx1250) — compile-verified, matrix operand order checked by reading, not run on hardware
//
#include <hip/hip_runtime.h>
#include <hip/hip_bf16.h>
#include <math.h>

// ---------------------------------------------------------------------------
// Problem constants (match reference)
// ---------------------------------------------------------------------------
#define B_  2
#define N_  2048
#define D_  2048
#define H_  16
#define HD_ 128

typedef __attribute__((ext_vector_type(16))) __bf16 v16bf;
typedef __attribute__((ext_vector_type(8)))  __bf16 v8bf;
typedef __attribute__((ext_vector_type(8)))  float  v8f;

// ---------------------------------------------------------------------------
// CDNA5 async global->LDS copy (ASYNCcnt-tracked), with safe fallback.
// Builtin signature (from hipcc diagnostic): first param is
// 'int __attribute__((vector_size(16))) addrspace(1)*' (global), second is the
// LDS destination; trailing args are immediate offset + cache policy.
// ---------------------------------------------------------------------------
#if defined(__AMDGCN__) && __has_builtin(__builtin_amdgcn_global_load_async_to_lds_b128) && __has_builtin(__builtin_amdgcn_s_wait_asynccnt)
#define USE_ASYNC_LDS 1
#else
#define USE_ASYNC_LDS 0
#endif

#if USE_ASYNC_LDS
typedef int v4i_vec __attribute__((vector_size(16)));
typedef __attribute__((address_space(1))) v4i_vec g_v4i;
typedef __attribute__((address_space(3))) v4i_vec l_v4i;
__device__ __forceinline__ void copy16_g2l(const void* g, void* l) {
  __builtin_amdgcn_global_load_async_to_lds_b128((g_v4i*)g, (l_v4i*)l, 0, 0);
}
__device__ __forceinline__ void copy16_wait() { __builtin_amdgcn_s_wait_asynccnt(0); }
#else
__device__ __forceinline__ void copy16_g2l(const void* g, void* l) {
  *(v8bf*)l = *(const v8bf*)g;
}
__device__ __forceinline__ void copy16_wait() {}
#endif

// ---------------------------------------------------------------------------
// Fragment loader (works for LDS or global bf16 sources, K-contiguous rows).
// CDNA5 16-bit A-layout (ISA 7.12.2): lane r(=lane&15) = row; elements 0..7 =
// K hi*8+0..7, elements 8..15 = K 16+hi*8+0..7, hi = lane>>4. B fragments use
// the symmetric N-major/K-striped form, so the same loader serves both.
// ---------------------------------------------------------------------------
__device__ __forceinline__ v16bf ld_frag(const __bf16* row0, int ld, int lane) {
  const int r  = lane & 15;
  const int hi = (lane >> 4) & 1;
  const __bf16* p = row0 + (size_t)r * ld;
  v8bf lo  = *(const v8bf*)(p + hi * 8);
  v8bf hi8 = *(const v8bf*)(p + 16 + hi * 8);
  return __builtin_shufflevector(lo, hi8, 0,1,2,3,4,5,6,7,8,9,10,11,12,13,14,15);
}

// ---------------------------------------------------------------------------
// f32 -> bf16 bulk convert (n multiple of 8)
// ---------------------------------------------------------------------------
__global__ __launch_bounds__(256)
void cvt_f32_bf16(const float* __restrict__ in, __bf16* __restrict__ out, int n)
{
  int i = (blockIdx.x * 256 + threadIdx.x) * 8;
  if (i + 8 <= n) {
#pragma unroll
    for (int j = 0; j < 8; ++j) out[i + j] = (__bf16)in[i + j];
  }
}

// ---------------------------------------------------------------------------
// GEMM: C = A @ W^T + bias.  A:[M,K] bf16, W:[N,K] bf16 row-major, C f32.
// blockIdx.z selects among 3 weight sets (fused QKV projection).
// Block tile 128x64, 8 waves, each wave 32x32 (4 WMMAs / K-step of 32).
// ---------------------------------------------------------------------------
__global__ __launch_bounds__(256)
void gemm_nt_bias(const __bf16* __restrict__ A,
                  const __bf16* __restrict__ W0, const __bf16* __restrict__ W1,
                  const __bf16* __restrict__ W2,
                  const float* __restrict__ b0, const float* __restrict__ b1,
                  const float* __restrict__ b2,
                  float* __restrict__ C0, float* __restrict__ C1,
                  float* __restrict__ C2,
                  int M, int Nn, int K)
{
  const __bf16* W = W0; const float* bias = b0; float* C = C0;
  if (blockIdx.z == 1) { W = W1; bias = b1; C = C1; }
  else if (blockIdx.z == 2) { W = W2; bias = b2; C = C2; }

  const int m0   = blockIdx.y * 128;
  const int n0   = blockIdx.x * 64;
  const int tid  = threadIdx.x;
  const int lane = tid & 31;
  const int wv   = tid >> 5;     // 0..7
  const int mw   = wv >> 1;      // 0..3 : 32-row strip
  const int nw   = wv & 1;       // 0..1 : 32-col strip

  __shared__ __align__(16) __bf16 sA[128 * 40];  // K-stride 40 keeps 16B align
  __shared__ __align__(16) __bf16 sW[64 * 40];

  v8f acc00, acc01, acc10, acc11;
#pragma unroll
  for (int i = 0; i < 8; ++i) { acc00[i]=0.f; acc01[i]=0.f; acc10[i]=0.f; acc11[i]=0.f; }

  for (int k0 = 0; k0 < K; k0 += 32) {
    __syncthreads();
    // A tile 128x32 bf16 = 512 16B chunks; 2 chunks per thread.
#pragma unroll
    for (int i = 0; i < 2; ++i) {
      int c = tid + i * 256;
      int row = c >> 2, col8 = (c & 3) * 8;
      copy16_g2l(A + (size_t)(m0 + row) * K + k0 + col8, sA + row * 40 + col8);
    }
    // W tile 64x32 bf16 = 256 16B chunks; 1 chunk per thread.
    {
      int row = tid >> 2, col8 = (tid & 3) * 8;
      copy16_g2l(W + (size_t)(n0 + row) * K + k0 + col8, sW + row * 40 + col8);
    }
    if (k0 + 32 < K) {  // prefetch next tiles -> global_prefetch_b8
      __builtin_prefetch(A + (size_t)(m0 + (tid >> 1)) * K + k0 + 32, 0, 0);
      if (tid < 64) __builtin_prefetch(W + (size_t)(n0 + tid) * K + k0 + 32, 0, 0);
    }
    copy16_wait();
    __syncthreads();

    v16bf a0  = ld_frag(sA + (mw * 32)      * 40, 40, lane);
    v16bf a1  = ld_frag(sA + (mw * 32 + 16) * 40, 40, lane);
    v16bf bb0 = ld_frag(sW + (nw * 32)      * 40, 40, lane);
    v16bf bb1 = ld_frag(sW + (nw * 32 + 16) * 40, 40, lane);

    acc00 = __builtin_amdgcn_wmma_f32_16x16x32_bf16(false, a0, false, bb0, (short)0, acc00, false, false);
    acc01 = __builtin_amdgcn_wmma_f32_16x16x32_bf16(false, a0, false, bb1, (short)0, acc01, false, false);
    acc10 = __builtin_amdgcn_wmma_f32_16x16x32_bf16(false, a1, false, bb0, (short)0, acc10, false, false);
    acc11 = __builtin_amdgcn_wmma_f32_16x16x32_bf16(false, a1, false, bb1, (short)0, acc11, false, false);
  }

  // C layout: VGPR slot r -> row r (lanes 0-15) / row r+8 (lanes 16-31), col = lane&15
  const int hi = (lane >> 4) & 1;
  const int cn = lane & 15;
#pragma unroll
  for (int r = 0; r < 8; ++r) {
    int row = r + 8 * hi;
    {
      int gm = m0 + mw * 32 + row, gn = n0 + nw * 32 + cn;
      C[(size_t)gm * Nn + gn]      = acc00[r] + bias[gn];
      C[(size_t)gm * Nn + gn + 16] = acc01[r] + bias[gn + 16];
    }
    {
      int gm = m0 + mw * 32 + 16 + row, gn = n0 + nw * 32 + cn;
      C[(size_t)gm * Nn + gn]      = acc10[r] + bias[gn];
      C[(size_t)gm * Nn + gn + 16] = acc11[r] + bias[gn + 16];
    }
  }
}

// ---------------------------------------------------------------------------
// RMSNorm + RoPE: f32 in, bf16 out. grid = (B*N rows, 2: q/k).
// freqs: [N,1,HD/2,2] (cos,sin).
// ---------------------------------------------------------------------------
__global__ __launch_bounds__(256)
void rmsnorm_rope(const float* __restrict__ yq, const float* __restrict__ yk,
                  __bf16* __restrict__ qb, __bf16* __restrict__ kb,
                  const float* __restrict__ gq, const float* __restrict__ gk,
                  const float* __restrict__ freqs)
{
  const float* xin = (blockIdx.y == 0) ? yq : yk;
  __bf16* xout     = (blockIdx.y == 0) ? qb : kb;
  const float* g   = (blockIdx.y == 0) ? gq : gk;
  const int row = blockIdx.x;
  const int tid = threadIdx.x;
  const float* x = xin + (size_t)row * D_;
  __bf16* o      = xout + (size_t)row * D_;

  __shared__ float red[256];
  float s = 0.f;
  for (int i = tid; i < D_; i += 256) { float vv = x[i]; s += vv * vv; }
  red[tid] = s;
  __syncthreads();
  for (int off = 128; off > 0; off >>= 1) {
    if (tid < off) red[tid] += red[tid + off];
    __syncthreads();
  }
  const float rms = rsqrtf(red[0] / (float)D_ + 1e-6f);
  const int pos = row % N_;

  for (int p = tid; p < D_ / 2; p += 256) {
    int h  = p >> 6;         // head = pair / 64
    int pp = p & 63;         // pair within head
    int i0 = h * HD_ + 2 * pp;
    float x0 = x[i0]     * rms * g[i0];
    float x1 = x[i0 + 1] * rms * g[i0 + 1];
    float c  = freqs[((size_t)pos * 64 + pp) * 2 + 0];
    float si = freqs[((size_t)pos * 64 + pp) * 2 + 1];
    o[i0]     = (__bf16)(x0 * c  - x1 * si);
    o[i0 + 1] = (__bf16)(x0 * si + x1 * c);
  }
}

// ---------------------------------------------------------------------------
// Flash attention, all-bf16 operands. Block = 4 waves, per (b,h,64 q-rows).
// Wave owns 16 q-rows; kv streamed in 32-row tiles via LDS.
// ---------------------------------------------------------------------------
__global__ __launch_bounds__(128)
void flash_attn(const __bf16* __restrict__ q, const __bf16* __restrict__ k,
                const __bf16* __restrict__ v, float* __restrict__ out)
{
  const int qb   = blockIdx.x;
  const int hh   = blockIdx.y;
  const int bb   = blockIdx.z;
  const int tid  = threadIdx.x;
  const int lane = tid & 31;
  const int wv   = tid >> 5;              // 0..3
  const int q0   = qb * 64 + wv * 16;
  const int hi   = (lane >> 4) & 1;
  const int cn   = lane & 15;
  const float scale = 0.08838834764831845f;   // 1/sqrt(128)

  __shared__ __align__(16) __bf16 sK [32 * 136];   // [kv][d], stride 136
  __shared__ __align__(16) __bf16 sVt[128 * 40];   // [d][kv], stride 40 (transposed)
  __shared__ __align__(16) __bf16 sP [4][16 * 32]; // per-wave P tile

  // Q tile 16x128 as 4 A-fragments, loaded once from global bf16.
  const __bf16* qrow0 = q + ((size_t)(bb * N_ + q0)) * D_ + hh * HD_;
  v16bf qf[4];
#pragma unroll
  for (int ks = 0; ks < 4; ++ks)
    qf[ks] = ld_frag(qrow0 + ks * 32, D_, lane);

  v8f o[8];
#pragma unroll
  for (int df = 0; df < 8; ++df) {
#pragma unroll
    for (int r = 0; r < 8; ++r) o[df][r] = 0.f;
  }
  float m_r[8], l_r[8];
#pragma unroll
  for (int r = 0; r < 8; ++r) { m_r[r] = -3.0e38f; l_r[r] = 0.f; }

  for (int kt = 0; kt < N_; kt += 32) {
    __syncthreads();
    const __bf16* kbase = k + ((size_t)(bb * N_ + kt)) * D_ + hh * HD_;
    const __bf16* vbase = v + ((size_t)(bb * N_ + kt)) * D_ + hh * HD_;
    // K tile 32x128 bf16 = 512 16B chunks; async copy, 4 per thread.
#pragma unroll
    for (int i = 0; i < 4; ++i) {
      int c = tid + i * 128;
      int row = c >> 4, col8 = (c & 15) * 8;
      copy16_g2l(kbase + (size_t)row * D_ + col8, sK + row * 136 + col8);
    }
    // V tile: vectorized global load, transposed scalar LDS scatter.
#pragma unroll
    for (int i = 0; i < 4; ++i) {
      int c = tid + i * 128;
      int row = c >> 4, col8 = (c & 15) * 8;
      v8bf vv = *(const v8bf*)(vbase + (size_t)row * D_ + col8);
#pragma unroll
      for (int j = 0; j < 8; ++j) sVt[(col8 + j) * 40 + row] = vv[j];
    }
    copy16_wait();
    __syncthreads();

    // S = Q K^T : two 16x16 C-frags (kv cols 0-15, 16-31), K-dim = 128.
    v8f s0, s1;
#pragma unroll
    for (int r = 0; r < 8; ++r) { s0[r] = 0.f; s1[r] = 0.f; }
#pragma unroll
    for (int ks = 0; ks < 4; ++ks) {
      v16bf kf0 = ld_frag(sK + 0  * 136 + ks * 32, 136, lane);
      v16bf kf1 = ld_frag(sK + 16 * 136 + ks * 32, 136, lane);
      s0 = __builtin_amdgcn_wmma_f32_16x16x32_bf16(false, qf[ks], false, kf0, (short)0, s0, false, false);
      s1 = __builtin_amdgcn_wmma_f32_16x16x32_bf16(false, qf[ks], false, kf1, (short)0, s1, false, false);
    }

    // Online softmax. Reductions stay inside each 16-lane half (lanes 0-15
    // hold rows r, lanes 16-31 rows r+8 — matches C-fragment layout).
#pragma unroll
    for (int r = 0; r < 8; ++r) {
      float a0 = s0[r] * scale;
      float a1 = s1[r] * scale;
      float mx = fmaxf(a0, a1);
      mx = fmaxf(mx, __shfl_xor(mx, 1, 32));
      mx = fmaxf(mx, __shfl_xor(mx, 2, 32));
      mx = fmaxf(mx, __shfl_xor(mx, 4, 32));
      mx = fmaxf(mx, __shfl_xor(mx, 8, 32));
      float mnew  = fmaxf(m_r[r], mx);
      float alpha = __expf(m_r[r] - mnew);
      float p0 = __expf(a0 - mnew);
      float p1 = __expf(a1 - mnew);
      float rs = p0 + p1;
      rs += __shfl_xor(rs, 1, 32);
      rs += __shfl_xor(rs, 2, 32);
      rs += __shfl_xor(rs, 4, 32);
      rs += __shfl_xor(rs, 8, 32);
      l_r[r] = l_r[r] * alpha + rs;
      m_r[r] = mnew;
#pragma unroll
      for (int df = 0; df < 8; ++df) o[df][r] *= alpha;
      // C-layout -> LDS (wave-local; LDS ops are in-order per wave)
      sP[wv][(r + 8 * hi) * 32 + cn]      = (__bf16)p0;
      sP[wv][(r + 8 * hi) * 32 + 16 + cn] = (__bf16)p1;
    }

    // O += P V : P as A-frag (16x32), V^T rows are d, elems stripe kv.
    v16bf pa = ld_frag(&sP[wv][0], 32, lane);
#pragma unroll
    for (int df = 0; df < 8; ++df) {
      v16bf vb = ld_frag(sVt + (df * 16) * 40, 40, lane);
      o[df] = __builtin_amdgcn_wmma_f32_16x16x32_bf16(false, pa, false, vb, (short)0, o[df], false, false);
    }
  }

  float* obase = out + ((size_t)(bb * N_)) * D_ + hh * HD_;
#pragma unroll
  for (int r = 0; r < 8; ++r) {
    float inv = 1.0f / l_r[r];
    int row = q0 + r + 8 * hi;
#pragma unroll
    for (int df = 0; df < 8; ++df)
      obase[(size_t)row * D_ + df * 16 + cn] = o[df][r] * inv;
  }
}

// ---------------------------------------------------------------------------
// Host side
// ---------------------------------------------------------------------------
extern "C" void kernel_launch(void* const* d_in, const int* in_sizes, int n_in,
                              void* d_out, int out_size, void* d_ws, size_t ws_size,
                              hipStream_t stream)
{
  const float* x     = (const float*)d_in[0];
  const float* freqs = (const float*)d_in[1];
  const float* wq    = (const float*)d_in[2];
  const float* bq    = (const float*)d_in[3];
  const float* wk    = (const float*)d_in[4];
  const float* bk    = (const float*)d_in[5];
  const float* wv    = (const float*)d_in[6];
  const float* bv    = (const float*)d_in[7];
  const float* wo    = (const float*)d_in[8];
  const float* bo    = (const float*)d_in[9];
  const float* gq    = (const float*)d_in[10];
  const float* gk    = (const float*)d_in[11];
  float* outp = (float*)d_out;

  const int M = B_ * N_;                      // 4096
  const size_t actN = (size_t)M * D_;         // 8M elems
  const size_t wN   = (size_t)D_ * D_;        // 4M elems

  char* p = (char*)d_ws;
  float*  yq   = (float*)p;                p += actN * 4;   // f32 activations
  float*  yk   = (float*)p;                p += actN * 4;
  float*  yv   = (float*)p;                p += actN * 4;
  __bf16* xbf  = (__bf16*)p;               p += actN * 2;   // bf16 operands
  __bf16* wqb  = (__bf16*)p;               p += wN * 2;
  __bf16* wkb  = (__bf16*)p;               p += wN * 2;
  __bf16* wvb  = (__bf16*)p;               p += wN * 2;
  __bf16* wob  = (__bf16*)p;               p += wN * 2;
  __bf16* qb   = (__bf16*)p;               p += actN * 2;
  __bf16* kb   = (__bf16*)p;               p += actN * 2;
  __bf16* vb   = (__bf16*)p;               p += actN * 2;
  float*  attn   = yq;                 // f32 q dead after rmsnorm -> alias
  __bf16* attnbf = xbf;                // x_bf dead after QKV GEMM -> alias

  // 0) One-time bf16 conversions of GEMM operands.
  cvt_f32_bf16<<<dim3(actN / (8 * 256)), 256, 0, stream>>>(x,  xbf, (int)actN);
  cvt_f32_bf16<<<dim3(wN   / (8 * 256)), 256, 0, stream>>>(wq, wqb, (int)wN);
  cvt_f32_bf16<<<dim3(wN   / (8 * 256)), 256, 0, stream>>>(wk, wkb, (int)wN);
  cvt_f32_bf16<<<dim3(wN   / (8 * 256)), 256, 0, stream>>>(wv, wvb, (int)wN);
  cvt_f32_bf16<<<dim3(wN   / (8 * 256)), 256, 0, stream>>>(wo, wob, (int)wN);

  // 1) Fused Q/K/V projections (z selects weight set), f32 accum/out.
  gemm_nt_bias<<<dim3(D_ / 64, M / 128, 3), 256, 0, stream>>>(
      xbf, wqb, wkb, wvb, bq, bk, bv, yq, yk, yv, M, D_, D_);

  // 2) RMSNorm + RoPE on q,k (f32 in, bf16 out); v: plain bf16 convert.
  rmsnorm_rope<<<dim3(M, 2), 256, 0, stream>>>(yq, yk, qb, kb, gq, gk, freqs);
  cvt_f32_bf16<<<dim3(actN / (8 * 256)), 256, 0, stream>>>(yv, vb, (int)actN);

  // 3) Flash attention -> attn (f32), then bf16 for the output GEMM.
  flash_attn<<<dim3(N_ / 64, H_, B_), 128, 0, stream>>>(qb, kb, vb, attn);
  cvt_f32_bf16<<<dim3(actN / (8 * 256)), 256, 0, stream>>>(attn, attnbf, (int)actN);

  // 4) Output projection -> d_out.
  gemm_nt_bias<<<dim3(D_ / 64, M / 128, 1), 256, 0, stream>>>(
      attnbf, wob, wob, wob, bo, bo, bo, outp, outp, outp, M, D_, D_);
}
